// Autoencoder_90125593740123
// MI455X (gfx1250) — matrix operand order repaired and checked by reading, not compiled, not run on hardware
//
#include <hip/hip_runtime.h>

// ---------------------------------------------------------------------------
// TopK sparse autoencoder forward for MI455X (gfx1250, wave32).
//   B=4096, N_IN=4096, N_LAT=32768, K(topk)=64
// Compute plan (bf16 WMMA, f32 accum, async-to-LDS double buffering):
//   encode GEMM  (1.1 TFLOP)  : full_latents = relu(xb @ enc + lat_bias)
//   radix-select top-64/row   : latents (dense) + compact (idx,val)
//   sparse decode (2 GFLOP)   : reconstructed = sum val*dec[idx] + pre_bias
//   decode GEMM  (1.1 TFLOP)  : reconstructed_full = full_latents @ dec + pre_bias
// ---------------------------------------------------------------------------

typedef __attribute__((ext_vector_type(16))) __bf16 v16bf;
typedef __attribute__((ext_vector_type(8)))  float  v8f;

union Frag { v16bf v; uint4 q[2]; };

__device__ __forceinline__ unsigned short f2bf(float f) {
    unsigned u = __float_as_uint(f);
    unsigned r = u + 0x7FFFu + ((u >> 16) & 1u);   // round-to-nearest-even
    return (unsigned short)(r >> 16);
}

// ---- CDNA5 async global->LDS copy (ASYNCcnt), inline asm (toolchain-portable)
__device__ __forceinline__ void async_ld_b128(void* ldsp, const void* g) {
    unsigned lds_off = (unsigned)(unsigned long long)ldsp;     // addr[31:0] = LDS byte addr
    unsigned long long ga = (unsigned long long)g;
    asm volatile("global_load_async_to_lds_b128 %0, %1, off"
                 :: "v"(lds_off), "v"(ga) : "memory");
}
__device__ __forceinline__ void wait_async0() {
    asm volatile("s_wait_asynccnt 0x0" ::: "memory");
}

// ---------------- fused f32->bf16 + transpose: out[C][R] = bf16(in[R][C]) ---
__global__ __launch_bounds__(256) void cvtT_kernel(const float* __restrict__ in,
                                                   unsigned short* __restrict__ out,
                                                   int R, int C) {
    __shared__ float t[32][33];
    const long long r0 = (long long)blockIdx.y * 32;
    const long long c0 = (long long)blockIdx.x * 32;
    const int tx = threadIdx.x & 31, ty = threadIdx.x >> 5;    // ty: 0..7
#pragma unroll
    for (int i = 0; i < 4; ++i)
        t[ty + 8 * i][tx] = in[(r0 + ty + 8 * i) * C + c0 + tx];
    __syncthreads();
#pragma unroll
    for (int i = 0; i < 4; ++i)
        out[(c0 + ty + 8 * i) * R + r0 + tx] = f2bf(t[tx][ty + 8 * i]);
}

// ---------------- xb = bf16(x - pre_bias) ----------------------------------
__global__ __launch_bounds__(256) void prep_x_kernel(const float4* __restrict__ x,
                                                     const float4* __restrict__ bias,
                                                     ushort4* __restrict__ out,
                                                     long long n4, int nin4) {
    long long i = (long long)blockIdx.x * blockDim.x + threadIdx.x;
    long long stride = (long long)gridDim.x * blockDim.x;
    for (; i < n4; i += stride) {
        float4 f = x[i];
        float4 b = bias[i % nin4];
        ushort4 o;
        o.x = f2bf(f.x - b.x); o.y = f2bf(f.y - b.y);
        o.z = f2bf(f.z - b.z); o.w = f2bf(f.w - b.w);
        out[i] = o;
    }
}

// ---------------- NT bf16 GEMM: C[M,N] = A[M,K] @ BT[N,K]^T ----------------
// v_wmma_f32_16x16x32_bf16; async double-buffered LDS staging.
// mode 0: v = relu(acc + bias[n]); out0=f32, out1=bf16
// mode 1: v = acc + bias[n];       out0=f32
__global__ __launch_bounds__(256) void gemm_bf16_nt_kernel(
        const unsigned short* __restrict__ A,
        const unsigned short* __restrict__ BT,
        float* __restrict__ out0,
        unsigned short* __restrict__ out1,
        const float* __restrict__ bias,
        int M, int N, int K, int mode) {
    // [buf][A=0/B=1][128 rows x pitch 40]  => 4 * 10240 B = 40 KB
    __shared__ __align__(16) unsigned short lds[2][2][128 * 40];

    const int tid  = threadIdx.x;
    const int lane = tid & 31;
    const int wave = tid >> 5;
    const int wm   = wave & 3;      // wave grid: 4 along M, 2 along N
    const int wn   = wave >> 2;
    const int ln   = lane & 15;
    const int hi   = lane >> 4;
    const int m0   = blockIdx.y * 128;
    const int n0   = blockIdx.x * 128;

    v8f acc[2][4];
#pragma unroll
    for (int i = 0; i < 2; ++i)
#pragma unroll
        for (int j = 0; j < 4; ++j)
            acc[i][j] = (v8f){0.f, 0.f, 0.f, 0.f, 0.f, 0.f, 0.f, 0.f};

    // stage tile (128x32 of A and of BT) into buffer `buf` via async DMA:
    // each thread fires 4 x global_load_async_to_lds_b128 (per-lane 16B copies).
    auto stage = [&](int buf, int k0) {
#pragma unroll
        for (int s = tid; s < 512; s += 256) {
            int r = s >> 2, c8 = (s & 3) * 8;
            async_ld_b128(&lds[buf][0][r * 40 + c8],
                          A  + (long long)(m0 + r) * K + k0 + c8);
            async_ld_b128(&lds[buf][1][r * 40 + c8],
                          BT + (long long)(n0 + r) * K + k0 + c8);
        }
    };

    const int nk = K >> 5;
    stage(0, 0);
    for (int t = 0; t < nk; ++t) {
        const int buf = t & 1;
        wait_async0();          // tile t landed (this wave's copies)
        __syncthreads();        // everyone's copies of tile t visible
        if (t + 1 < nk) stage(buf ^ 1, (t + 1) << 5);   // overlaps compute(t)

        // A frag: lane m=ln; elem j<8 -> k=hi*8+j ; j>=8 -> k=16+hi*8+(j-8)
        Frag a[2], b[4];
#pragma unroll
        for (int i = 0; i < 2; ++i) {
            int m = wm * 32 + i * 16 + ln;
            a[i].q[0] = *(const uint4*)(&lds[buf][0][m * 40 + hi * 8]);
            a[i].q[1] = *(const uint4*)(&lds[buf][0][m * 40 + 16 + hi * 8]);
        }
        // B frag: lane n=ln; elem j<8 -> k=hi*8+j ; j>=8 -> k=16+hi*8+(j-8)
#pragma unroll
        for (int j = 0; j < 4; ++j) {
            int n = wn * 64 + j * 16 + ln;
            b[j].q[0] = *(const uint4*)(&lds[buf][1][n * 40 + hi * 8]);
            b[j].q[1] = *(const uint4*)(&lds[buf][1][n * 40 + 16 + hi * 8]);
        }
#pragma unroll
        for (int i = 0; i < 2; ++i)
#pragma unroll
            for (int j = 0; j < 4; ++j)
                acc[i][j] = __builtin_amdgcn_wmma_f32_16x16x32_bf16(
                    false, a[i].v, false, b[j].v, (short)0, acc[i][j], false, false);
        __syncthreads();        // all reads of buf done before it is re-staged
    }

    // epilogue: D vgpr r -> row = r + 8*hi, col = ln
#pragma unroll
    for (int i = 0; i < 2; ++i) {
#pragma unroll
        for (int j = 0; j < 4; ++j) {
            int mbase = m0 + wm * 32 + i * 16 + hi * 8;
            int n = n0 + wn * 64 + j * 16 + ln;
            float bv = bias ? bias[n] : 0.f;
#pragma unroll
            for (int r = 0; r < 8; ++r) {
                long long idx = (long long)(mbase + r) * N + n;
                float v = acc[i][j][r] + bv;
                if (mode == 0) {
                    v = v > 0.f ? v : 0.f;
                    out0[idx] = v;
                    out1[idx] = f2bf(v);
                } else {
                    out0[idx] = v;
                }
            }
        }
    }
}

// ---------------- per-row top-k via 4-pass radix select --------------------
// values are relu'd (>=0) so float-bit order == unsigned order.
__global__ __launch_bounds__(256) void topk_kernel(const float* __restrict__ fl,
                                                   float* __restrict__ latents,
                                                   int* __restrict__ idxbuf,
                                                   float* __restrict__ valbuf,
                                                   const int* __restrict__ kp) {
    __shared__ unsigned hist[256];
    __shared__ unsigned s_pref, s_need, s_eq, s_out;
    const int tid = threadIdx.x;
    const long long row = blockIdx.x;
    const float* r = fl + row * 32768LL;

    int k = *kp;
    if (k < 1) k = 1;
    if (k > 64) k = 64;   // compact buffers are stride-64

    if (tid == 0) { s_pref = 0; s_need = (unsigned)k; }
    __syncthreads();

    for (int p = 0; p < 4; ++p) {
        hist[tid] = 0;
        __syncthreads();
        unsigned pref = s_pref, need = s_need;
        int sh = 24 - 8 * p;
        for (int e = tid; e < 32768; e += 256) {
            unsigned u = __float_as_uint(r[e]);
            if (p == 0 || (u >> (32 - 8 * p)) == pref)
                atomicAdd(&hist[(u >> sh) & 255u], 1u);
        }
        __syncthreads();
        if (tid == 0) {
            unsigned cum = 0;
            for (int d = 255; d >= 0; --d) {
                unsigned h = hist[d];
                if (cum + h >= need) { s_pref = (pref << 8) | (unsigned)d; s_need = need - cum; break; }
                cum += h;
            }
        }
        __syncthreads();
    }

    const unsigned T = s_pref;       // bit pattern of k-th largest
    const unsigned quota = s_need;   // how many ==T to keep
    if (tid == 0) { s_eq = 0; s_out = 0; }
    __syncthreads();

    for (int e = tid; e < 32768; e += 256) {
        float v = r[e];
        unsigned u = __float_as_uint(v);
        float o = 0.f;
        bool sel = false;
        if (u > T) { o = v; sel = true; }
        else if (u == T) {
            unsigned slot = atomicAdd(&s_eq, 1u);
            if (slot < quota) { o = v; sel = true; }
        }
        latents[row * 32768LL + e] = o;
        if (sel) {
            unsigned pos = atomicAdd(&s_out, 1u);
            idxbuf[row * 64 + pos] = e;
            valbuf[row * 64 + pos] = o;
        }
    }
}

// ---------------- sparse decode (f32 decoder, 64 nnz/row) ------------------
__global__ __launch_bounds__(256) void sparse_decode_kernel(
        const int* __restrict__ idxbuf, const float* __restrict__ valbuf,
        const float* __restrict__ dec, const float* __restrict__ pre_bias,
        float* __restrict__ recon, const int* __restrict__ kp) {
    int k = *kp; if (k < 1) k = 1; if (k > 64) k = 64;
    const long long row = blockIdx.x;
    const int col0 = threadIdx.x * 16;     // 256 threads x 16 cols = 4096
    float acc[16];
#pragma unroll
    for (int c = 0; c < 16; ++c) acc[c] = 0.f;

    const int*   ib = idxbuf + row * 64;
    const float* vb = valbuf + row * 64;
    for (int j = 0; j < k; ++j) {
        int id = ib[j];
        float v = vb[j];
        const float4* dr = (const float4*)(dec + (long long)id * 4096 + col0);
#pragma unroll
        for (int q = 0; q < 4; ++q) {
            float4 h = dr[q];
            acc[q * 4 + 0] += v * h.x;
            acc[q * 4 + 1] += v * h.y;
            acc[q * 4 + 2] += v * h.z;
            acc[q * 4 + 3] += v * h.w;
        }
    }
#pragma unroll
    for (int c = 0; c < 16; ++c)
        recon[row * 4096LL + col0 + c] = acc[c] + pre_bias[col0 + c];
}

// ---------------------------------------------------------------------------
extern "C" void kernel_launch(void* const* d_in, const int* in_sizes, int n_in,
                              void* d_out, int out_size, void* d_ws, size_t ws_size,
                              hipStream_t stream) {
    (void)in_sizes; (void)n_in; (void)out_size; (void)ws_size;
    const int B = 4096, N_IN = 4096, N_LAT = 32768;

    const float* x        = (const float*)d_in[0];
    const float* pre_bias = (const float*)d_in[1];
    const float* enc      = (const float*)d_in[2];
    const float* lat_bias = (const float*)d_in[3];
    const float* dec      = (const float*)d_in[4];
    const int*   kp       = (const int*)d_in[5];

    float* out          = (float*)d_out;
    float* recon        = out;                                   // [B,N_IN]
    float* latents      = out + (long long)B * N_IN;             // [B,N_LAT]
    float* recon_full   = latents + (long long)B * N_LAT;        // [B,N_IN]
    float* full_latents = recon_full + (long long)B * N_IN;      // [B,N_LAT]

    char* ws = (char*)d_ws;
    unsigned short* encT_bf = (unsigned short*)(ws);                   // [N_LAT][N_IN] 268 MB
    unsigned short* decT_bf = (unsigned short*)(ws + 268435456LL);     // [N_IN][N_LAT] 268 MB
    unsigned short* fl_bf   = (unsigned short*)(ws + 536870912LL);     // [B][N_LAT]    268 MB
    unsigned short* xb      = (unsigned short*)(ws + 805306368LL);     // [B][N_IN]    33.5 MB
    int*            idxbuf  = (int*)  (ws + 838860800LL);              // [B][64]
    float*          valbuf  = (float*)(ws + 839909376LL);              // [B][64]

    // 1) fused convert+transpose of weights; prepare xb = bf16(x - pre_bias)
    cvtT_kernel<<<dim3(N_LAT / 32, N_IN / 32), 256, 0, stream>>>(enc, encT_bf, N_IN, N_LAT);
    cvtT_kernel<<<dim3(N_IN / 32, N_LAT / 32), 256, 0, stream>>>(dec, decT_bf, N_LAT, N_IN);
    prep_x_kernel<<<16384, 256, 0, stream>>>((const float4*)x, (const float4*)pre_bias,
                                             (ushort4*)xb, (long long)B * N_IN / 4, N_IN / 4);

    // 2) encode GEMM: full_latents = relu(xb @ enc + lat_bias)  (+ bf16 copy)
    gemm_bf16_nt_kernel<<<dim3(N_LAT / 128, B / 128), 256, 0, stream>>>(
        xb, encT_bf, full_latents, fl_bf, lat_bias, B, N_LAT, N_IN, 0);

    // 3) per-row top-k: dense latents + compact (idx,val) lists
    topk_kernel<<<B, 256, 0, stream>>>(full_latents, latents, idxbuf, valbuf, kp);

    // 4) sparse decode: reconstructed = latents @ dec + pre_bias (64 nnz/row)
    sparse_decode_kernel<<<B, 256, 0, stream>>>(idxbuf, valbuf, dec, pre_bias, recon, kp);

    // 5) dense decode GEMM: reconstructed_full = full_latents @ dec + pre_bias
    gemm_bf16_nt_kernel<<<dim3(N_IN / 128, B / 128), 256, 0, stream>>>(
        fl_bf, decT_bf, recon_full, nullptr, pre_bias, B, N_IN, N_LAT, 1);
}
